// GraphUpSample_64759516889791
// MI455X (gfx1250) — matrix operand order repaired and checked by reading, not compiled, not run on hardware
//
#include <hip/hip_runtime.h>
#include <stdint.h>

// Problem constants (from reference)
#define BATCH   32
#define CIN     512
#define F_DIM   128
#define NNODES  16
#define COUT    512
#define KW      2
#define NPRIME  (COUT*KW)          // 1024
#define XSTRC   (F_DIM*NNODES)     // 2048: x stride for c
#define OSTRF   (NNODES*KW)        // 32:   out stride for f

#define KTILE   32
#define WG_M    128
#define WG_N    128
#define KSTEPS  (CIN/KTILE)        // 16
#define LDS_STR 36                 // bf16 elems per row (32 + 4 pad) -> 72B rows, bank-friendly

typedef __attribute__((ext_vector_type(16))) __bf16 v16bf;
typedef __attribute__((ext_vector_type(2)))  __bf16 v2bf;
typedef __attribute__((ext_vector_type(8)))  float  v8f;

union FragU { uint2 q[4]; v16bf v; };   // 32 bytes: one 16x32-bf16 A/B fragment per lane

// Let the backend choose the native convert (v_cvt_pk_bf16_f32 on gfx12-lineage HW);
// clang inlines an RNE software sequence if no HW instruction exists.
__device__ __forceinline__ uint32_t pack_bf16(float lo, float hi) {
  v2bf v;
  v[0] = (__bf16)lo;
  v[1] = (__bf16)hi;
  return __builtin_bit_cast(uint32_t, v);
}

__global__ __launch_bounds__(256)
void graph_upsample_wmma(const float* __restrict__ x,
                         const float* __restrict__ W,
                         const float* __restrict__ bias,
                         float* __restrict__ out) {
  // bf16 staging: A = [f][c'] (128x32), B^T = [j'][c'] (128x32), double buffered
  __shared__ uint16_t Alds[2][WG_M * LDS_STR];
  __shared__ uint16_t Blds[2][WG_N * LDS_STR];

  const int tid  = threadIdx.x;
  const int lane = tid & 31;
  const int wave = tid >> 5;       // 0..7
  const int wr   = wave >> 1;      // 0..3 : M direction (32 rows each)
  const int wc   = wave & 1;       // 0..1 : N direction (64 cols each)
  const int lr   = lane & 15;
  const int krow = lane >> 4;      // lane half selects K sub-range per ISA layout

  const int n  = blockIdx.z;       // node
  const int bb = blockIdx.y;       // batch (WG_M == F_DIM -> one batch per M tile)
  const int j0 = blockIdx.x * WG_N;

  const float* xbase = x + (size_t)bb * (CIN * XSTRC) + n;       // + c*2048 + f*16
  const float* wbase = W + (size_t)n  * (CIN * NPRIME) + j0;     // + c*1024 + j'

  // Global->reg staging map: 256 threads cover a 128x32 tile (row = tid&127, c half = tid>>7)
  const int arow = tid & 127;              // f for A, j' for B
  const int acol = (tid >> 7) << 4;        // 0 or 16

  float aR[16], bR[16];

  auto load_globals = [&](int ks) {
    const float* xp = xbase + (size_t)(ks * KTILE + acol) * XSTRC + arow * NNODES;
    const float* wp = wbase + (size_t)(ks * KTILE + acol) * NPRIME + arow;
#pragma unroll
    for (int i = 0; i < 16; ++i) aR[i] = xp[(size_t)i * XSTRC];
#pragma unroll
    for (int i = 0; i < 16; ++i) bR[i] = wp[(size_t)i * NPRIME];
    if (ks + 1 < KSTEPS) {   // warm caches for the tile after next
      __builtin_prefetch(xp + (size_t)KTILE * XSTRC, 0, 1);
      __builtin_prefetch(wp + (size_t)KTILE * NPRIME, 0, 1);
    }
  };

  auto store_lds = [&](int buf) {
    uint2* ap = (uint2*)&Alds[buf][arow * LDS_STR + acol];   // 72*arow + {0,32} bytes: 8B aligned
    uint2* bp = (uint2*)&Blds[buf][arow * LDS_STR + acol];
#pragma unroll
    for (int i = 0; i < 4; ++i) {
      uint2 qa, qb;
      qa.x = pack_bf16(aR[4*i+0], aR[4*i+1]); qa.y = pack_bf16(aR[4*i+2], aR[4*i+3]);
      qb.x = pack_bf16(bR[4*i+0], bR[4*i+1]); qb.y = pack_bf16(bR[4*i+2], bR[4*i+3]);
      ap[i] = qa;
      bp[i] = qb;
    }
  };

  v8f acc[2][4];
#pragma unroll
  for (int mi = 0; mi < 2; ++mi)
#pragma unroll
    for (int nj = 0; nj < 4; ++nj) acc[mi][nj] = (v8f){0.f,0.f,0.f,0.f,0.f,0.f,0.f,0.f};

  load_globals(0);
  store_lds(0);

  for (int ks = 0; ks < KSTEPS; ++ks) {
    __syncthreads();
    if (ks + 1 < KSTEPS) load_globals(ks + 1);   // VMEM in flight during WMMA

    const int buf = ks & 1;
    v16bf afrag[2], bfrag[4];
    // A fragment (16x32 bf16): lane = row m; K chunks [kb, kb+8) and [kb+16, kb+24), kb = krow*8
#pragma unroll
    for (int mi = 0; mi < 2; ++mi) {
      const uint16_t* rp = &Alds[buf][(wr * 32 + mi * 16 + lr) * LDS_STR];
      FragU fa;
      const uint2* p0 = (const uint2*)(rp + krow * 8);
      const uint2* p1 = (const uint2*)(rp + krow * 8 + 16);
      fa.q[0] = p0[0]; fa.q[1] = p0[1];
      fa.q[2] = p1[0]; fa.q[3] = p1[1];
      afrag[mi] = fa.v;
    }
    // B fragment (32x16 bf16): lane = col j; K chunks [kb, kb+8) and [kb+16, kb+24), kb = krow*8 (transposed LDS)
#pragma unroll
    for (int nj = 0; nj < 4; ++nj) {
      const uint16_t* rp = &Blds[buf][(wc * 64 + nj * 16 + lr) * LDS_STR + krow * 8];
      FragU fb;
      const uint2* p = (const uint2*)rp;
      fb.q[0] = p[0]; fb.q[1] = p[1]; fb.q[2] = p[4]; fb.q[3] = p[5];
      bfrag[nj] = fb.v;
    }
#pragma unroll
    for (int mi = 0; mi < 2; ++mi)
#pragma unroll
      for (int nj = 0; nj < 4; ++nj)
        acc[mi][nj] = __builtin_amdgcn_wmma_f32_16x16x32_bf16(
            false, afrag[mi], false, bfrag[nj],
            (short)0, acc[mi][nj], false, false);

    if (ks + 1 < KSTEPS) store_lds((ks + 1) & 1);  // writes the *other* buffer
  }

  // Epilogue: D layout = VGPR i -> row (i + 8*krow), col = lr. Add bias[n,o], scatter f32.
#pragma unroll
  for (int mi = 0; mi < 2; ++mi) {
#pragma unroll
    for (int nj = 0; nj < 4; ++nj) {
      const int jg = j0 + wc * 64 + nj * 16 + lr;   // j = o*2 + k
      const int o  = jg >> 1;
      const int kk = jg & 1;
      const float bv = bias[n * COUT + o];
      float* op = out + (size_t)bb * ((size_t)COUT * F_DIM * OSTRF)
                      + (size_t)o * (F_DIM * OSTRF)
                      + (size_t)n * KW + kk;
#pragma unroll
      for (int i = 0; i < 8; ++i) {
        const int f = wr * 32 + mi * 16 + krow * 8 + i;
        op[(size_t)f * OSTRF] = acc[mi][nj][i] + bv;
      }
    }
  }
}

extern "C" void kernel_launch(void* const* d_in, const int* in_sizes, int n_in,
                              void* d_out, int out_size, void* d_ws, size_t ws_size,
                              hipStream_t stream) {
  (void)in_sizes; (void)n_in; (void)out_size; (void)d_ws; (void)ws_size;
  const float* x    = (const float*)d_in[0];   // [32,512,128,16] f32
  const float* W    = (const float*)d_in[1];   // [16,512,512,2]  f32
  const float* bias = (const float*)d_in[2];   // [16,512]        f32
  float* out = (float*)d_out;                  // [32,512,128,32] f32

  dim3 grid(NPRIME / WG_N, BATCH, NNODES);     // (8, 32, 16) = 4096 WGs
  dim3 block(256);
  graph_upsample_wmma<<<grid, block, 0, stream>>>(x, W, bias, out);
}
